// ConformerMHA_73821897883874
// MI455X (gfx1250) — matrix operand order repaired and checked by reading, not compiled, not run on hardware
//
#include <hip/hip_runtime.h>
#include <hip/hip_bf16.h>

// ConformerMHA for MI455X (gfx1250): LN -> QKV gemm -> RoPE -> flash attention -> out gemm.
// All matmuls use v_wmma_f32_16x16x32_bf16 (wave32 WMMA), f32 accumulate.

typedef __attribute__((ext_vector_type(16))) __bf16 bf16x16;
typedef __attribute__((ext_vector_type(8)))  __bf16 bf16x8;
typedef __attribute__((ext_vector_type(8)))  float  f32x8;

namespace {
constexpr int kB   = 8;
constexpr int kT   = 2048;
constexpr int kD   = 512;
constexpr int kH   = 8;
constexpr int kDK  = 64;
constexpr int kNT  = kB * kT;   // 16384 tokens
constexpr int kQKV = 3 * kD;    // 1536
}

__device__ __forceinline__ f32x8 zero8() {
  f32x8 z;
  #pragma unroll
  for (int i = 0; i < 8; ++i) z[i] = 0.0f;
  return z;
}

// A-fragment (16x32 bf16), row-major source. Lane l: row = l&15.
// elements 0..7  = K (16*(l>>4) + j), elements 8..15 = K (16*(l>>4) + 8 + j).
__device__ __forceinline__ bf16x16 load_a_frag(const __bf16* rowp, int half) {
  bf16x8 lo = *(const bf16x8*)(rowp + 16 * half);
  bf16x8 hi = *(const bf16x8*)(rowp + 16 * half + 8);
  bf16x16 r;
  #pragma unroll
  for (int j = 0; j < 8; ++j) { r[j] = lo[j]; r[j + 8] = hi[j]; }
  return r;
}

__device__ __forceinline__ f32x8 wmma_bf16(bf16x16 a, bf16x16 b, f32x8 c) {
  return __builtin_amdgcn_wmma_f32_16x16x32_bf16(false, a, false, b, (short)0, c,
                                                 false, false);
}

// Reductions across the 16 lanes of each half-wave (wave32): xor masks 1,2,4,8
// never cross the 16-lane boundary, so each half reduces independently.
__device__ __forceinline__ float redmax16(float v) {
  #pragma unroll
  for (int m = 1; m < 16; m <<= 1) v = fmaxf(v, __shfl_xor(v, m, 32));
  return v;
}
__device__ __forceinline__ float redsum16(float v) {
  #pragma unroll
  for (int m = 1; m < 16; m <<= 1) v += __shfl_xor(v, m, 32);
  return v;
}

// ---------------------------------------------------------------------------
// 1) Weights -> bf16, transposed so WMMA B-fragments are contiguous loads.
//    wqkvT[n][k] = w_qkv[k][n] (1536x512), woT[n][k] = w_o[k][n] (512x512).
__global__ void prep_weights_kernel(const float* __restrict__ w_qkv,
                                    const float* __restrict__ w_o,
                                    __bf16* __restrict__ wqkvT,
                                    __bf16* __restrict__ woT) {
  const int i = blockIdx.x * 256 + threadIdx.x;
  const int NQ = kQKV * kD;          // 786432
  if (i < NQ) {
    const int n = i / kD, k = i % kD;
    wqkvT[i] = (__bf16)w_qkv[(long)k * kQKV + n];
  } else {
    const int j = i - NQ;            // j < 512*512
    const int n = j / kD, k = j % kD;
    woT[j] = (__bf16)w_o[(long)k * kD + n];
  }
}

// ---------------------------------------------------------------------------
// 2) LayerNorm per token row (512 wide), write bf16 activations.
__global__ void ln_kernel(const float* __restrict__ x, const float* __restrict__ w,
                          const float* __restrict__ bb, __bf16* __restrict__ hout) {
  __shared__ float red[256];
  const int r = blockIdx.x, tid = threadIdx.x;
  const float* xr = x + (long)r * kD;
  const float a0 = xr[tid], a1 = xr[tid + 256];

  red[tid] = a0 + a1;
  __syncthreads();
  #pragma unroll
  for (int st = 128; st > 0; st >>= 1) {
    if (tid < st) red[tid] += red[tid + st];
    __syncthreads();
  }
  const float mu = red[0] * (1.0f / kD);
  __syncthreads();

  const float d0 = a0 - mu, d1 = a1 - mu;
  red[tid] = d0 * d0 + d1 * d1;
  __syncthreads();
  #pragma unroll
  for (int st = 128; st > 0; st >>= 1) {
    if (tid < st) red[tid] += red[tid + st];
    __syncthreads();
  }
  const float inv = rsqrtf(red[0] * (1.0f / kD) + 1e-5f);

  hout[(long)r * kD + tid]       = (__bf16)(d0 * inv * w[tid] + bb[tid]);
  hout[(long)r * kD + tid + 256] = (__bf16)(d1 * inv * w[tid + 256] + bb[tid + 256]);
}

// ---------------------------------------------------------------------------
// 3) QKV GEMM: [16384,512]bf16 @ [512,1536] -> bf16 [16384,1536] (+bias).
//    One wave per 16x64 output tile, K-loop of 16 WMMA steps x 4 N-tiles.
__global__ void qkv_gemm_kernel(const __bf16* __restrict__ A, const __bf16* __restrict__ BT,
                                const float* __restrict__ bias, __bf16* __restrict__ C) {
  const int lane = threadIdx.x, half = lane >> 4, lcol = lane & 15;
  const int m0 = blockIdx.x * 16, n0 = blockIdx.y * 64;
  f32x8 acc[4];
  #pragma unroll
  for (int n = 0; n < 4; ++n) acc[n] = zero8();
  const __bf16* arow = A + (long)(m0 + lcol) * kD;
  for (int kb = 0; kb < kD; kb += 32) {
    __builtin_prefetch(arow + kb + 32, 0, 1);
    bf16x16 a = load_a_frag(arow + kb, half);
    #pragma unroll
    for (int n = 0; n < 4; ++n) {
      bf16x16 bf = *(const bf16x16*)(BT + (long)(n0 + n * 16 + lcol) * kD + kb + 16 * half);
      acc[n] = wmma_bf16(a, bf, acc[n]);
    }
  }
  #pragma unroll
  for (int n = 0; n < 4; ++n)
    #pragma unroll
    for (int i = 0; i < 8; ++i) {
      const int row = m0 + i + 8 * half;
      const int col = n0 + n * 16 + lcol;
      C[(long)row * kQKV + col] = (__bf16)(acc[n][i] + bias[col]);
    }
}

// ---------------------------------------------------------------------------
// 4) RoPE on q,k + repack: q,k -> [B,H,T,64] bf16; v -> transposed [B,H,64,T].
//    One thread per (b,h,t,i), i in [0,32): handles dims i and i+32.
__global__ void rope_kernel(const __bf16* __restrict__ qkv, __bf16* __restrict__ q,
                            __bf16* __restrict__ k, __bf16* __restrict__ vT) {
  const long g = (long)blockIdx.x * 256 + threadIdx.x;   // < B*H*T*32
  const int i = (int)(g & 31);
  const long bht = g >> 5;
  const int t  = (int)(bht % kT);
  const int bh = (int)(bht / kT);
  const int h  = bh & 7, b = bh >> 3;
  const __bf16* row = qkv + ((long)b * kT + t) * kQKV;

  const float q1 = (float)row[h * kDK + i],        q2 = (float)row[h * kDK + i + 32];
  const float k1 = (float)row[kD + h * kDK + i],   k2 = (float)row[kD + h * kDK + i + 32];
  const float invf = __powf(10000.0f, -(float)i * (1.0f / 32.0f));
  float sn, cs;
  __sincosf((float)t * invf, &sn, &cs);

  const long qb = ((long)bh * kT + t) * kDK;
  q[qb + i]      = (__bf16)(q1 * cs - q2 * sn);
  q[qb + i + 32] = (__bf16)(q2 * cs + q1 * sn);
  k[qb + i]      = (__bf16)(k1 * cs - k2 * sn);
  k[qb + i + 32] = (__bf16)(k2 * cs + k1 * sn);

  const long vb = (long)bh * kDK * kT;
  vT[vb + (long)i * kT + t]        = row[2 * kD + h * kDK + i];
  vT[vb + (long)(i + 32) * kT + t] = row[2 * kD + h * kDK + i + 32];
}

// ---------------------------------------------------------------------------
// 5) Flash attention. One wave per (b,h, 16-row q tile). Streams k in chunks
//    of 32 columns: 4 WMMA for scores, online softmax (half-wave butterflies),
//    P transposed via LDS into A-fragment layout, 4 WMMA for P@V.
__global__ void attn_kernel(const __bf16* __restrict__ qm, const __bf16* __restrict__ km,
                            const __bf16* __restrict__ vtm,
                            const unsigned char* __restrict__ mask,
                            __bf16* __restrict__ outm) {
  __shared__ __align__(32) __bf16 pbuf[16 * 32];
  const int lane = threadIdx.x, half = lane >> 4, lcol = lane & 15;
  const int q0 = blockIdx.x * 16;
  const int bh = blockIdx.y;
  const int b = bh >> 3, h = bh & 7;
  const long hb = (long)bh * kT * kDK;   // q/k head base
  const long vb = (long)bh * kDK * kT;   // vT head base

  bf16x16 aq[2];
  #pragma unroll
  for (int kb = 0; kb < 2; ++kb)
    aq[kb] = load_a_frag(qm + hb + (long)(q0 + lcol) * kDK + kb * 32, half);

  f32x8 o[4];
  #pragma unroll
  for (int n = 0; n < 4; ++n) o[n] = zero8();
  float mrow[8], lrow[8];
  #pragma unroll
  for (int i = 0; i < 8; ++i) { mrow[i] = -__builtin_inff(); lrow[i] = 0.0f; }

  for (int kc = 0; kc < kT; kc += 32) {
    // --- scores for 16x32 chunk (two 16x16 sub-tiles) ---
    f32x8 s[2];
    #pragma unroll
    for (int sub = 0; sub < 2; ++sub) {
      const int t0 = kc + sub * 16;
      f32x8 acc = zero8();
      #pragma unroll
      for (int kb = 0; kb < 2; ++kb) {
        // k is [t,d] row-major == B^T for q@k^T: lane lcol -> key column t0+lcol
        bf16x16 bk = *(const bf16x16*)(km + hb + (long)(t0 + lcol) * kDK + kb * 32 + 16 * half);
        acc = wmma_bf16(aq[kb], bk, acc);
      }
      const bool msk = mask[b * kT + t0 + lcol] != 0;
      #pragma unroll
      for (int i = 0; i < 8; ++i)
        s[sub][i] = msk ? -__builtin_inff() : acc[i] * 0.125f;  // 1/sqrt(64)
    }
    // --- online softmax update (per output row) ---
    float f_[8];
    #pragma unroll
    for (int i = 0; i < 8; ++i) {
      const float cm = redmax16(fmaxf(s[0][i], s[1][i]));
      const float mn = fmaxf(mrow[i], cm);
      f_[i] = __expf(mrow[i] - mn);   // exp(-inf)=0 on first chunk
      mrow[i] = mn;
    }
    #pragma unroll
    for (int sub = 0; sub < 2; ++sub)
      #pragma unroll
      for (int i = 0; i < 8; ++i)
        s[sub][i] = __expf(s[sub][i] - mrow[i]);
    #pragma unroll
    for (int i = 0; i < 8; ++i)
      lrow[i] = lrow[i] * f_[i] + redsum16(s[0][i] + s[1][i]);
    #pragma unroll
    for (int n = 0; n < 4; ++n)
      #pragma unroll
      for (int i = 0; i < 8; ++i)
        o[n][i] *= f_[i];

    // --- transpose P (C-layout -> A-layout) through LDS ---
    __syncthreads();
    #pragma unroll
    for (int sub = 0; sub < 2; ++sub)
      #pragma unroll
      for (int i = 0; i < 8; ++i)
        pbuf[(i + 8 * half) * 32 + sub * 16 + lcol] = (__bf16)s[sub][i];
    __syncthreads();

    bf16x16 ap;
    {
      bf16x8 lo = *(const bf16x8*)(pbuf + lcol * 32 + 16 * half);
      bf16x8 hi = *(const bf16x8*)(pbuf + lcol * 32 + 16 * half + 8);
      #pragma unroll
      for (int j = 0; j < 8; ++j) { ap[j] = lo[j]; ap[j + 8] = hi[j]; }
    }
    // --- P @ V: vT is [d,t] row-major == B^T, lane lcol -> output dim ---
    #pragma unroll
    for (int n = 0; n < 4; ++n) {
      bf16x16 bv = *(const bf16x16*)(vtm + vb + (long)(n * 16 + lcol) * kT + kc + 16 * half);
      o[n] = wmma_bf16(ap, bv, o[n]);
    }
    __syncthreads();
  }

  #pragma unroll
  for (int n = 0; n < 4; ++n)
    #pragma unroll
    for (int i = 0; i < 8; ++i) {
      const int row = q0 + i + 8 * half;
      const int col = h * kDK + n * 16 + lcol;
      outm[((long)b * kT + row) * kD + col] = (__bf16)(o[n][i] / lrow[i]);
    }
}

// ---------------------------------------------------------------------------
// 6) Output projection: [16384,512]bf16 @ [512,512] -> f32 [16384,512] (+bias).
__global__ void out_gemm_kernel(const __bf16* __restrict__ A, const __bf16* __restrict__ BT,
                                const float* __restrict__ bias, float* __restrict__ C) {
  const int lane = threadIdx.x, half = lane >> 4, lcol = lane & 15;
  const int m0 = blockIdx.x * 16, n0 = blockIdx.y * 64;
  f32x8 acc[4];
  #pragma unroll
  for (int n = 0; n < 4; ++n) acc[n] = zero8();
  const __bf16* arow = A + (long)(m0 + lcol) * kD;
  for (int kb = 0; kb < kD; kb += 32) {
    __builtin_prefetch(arow + kb + 32, 0, 1);
    bf16x16 a = load_a_frag(arow + kb, half);
    #pragma unroll
    for (int n = 0; n < 4; ++n) {
      bf16x16 bf = *(const bf16x16*)(BT + (long)(n0 + n * 16 + lcol) * kD + kb + 16 * half);
      acc[n] = wmma_bf16(a, bf, acc[n]);
    }
  }
  #pragma unroll
  for (int n = 0; n < 4; ++n)
    #pragma unroll
    for (int i = 0; i < 8; ++i) {
      const int row = m0 + i + 8 * half;
      const int col = n0 + n * 16 + lcol;
      C[(long)row * kD + col] = acc[n][i] + bias[col];
    }
}

// ---------------------------------------------------------------------------
extern "C" void kernel_launch(void* const* d_in, const int* in_sizes, int n_in,
                              void* d_out, int out_size, void* d_ws, size_t ws_size,
                              hipStream_t stream) {
  (void)in_sizes; (void)n_in; (void)out_size; (void)ws_size;
  const float*         x     = (const float*)d_in[0];
  const unsigned char* mask  = (const unsigned char*)d_in[1];  // all-false bool
  const float*         ln_w  = (const float*)d_in[2];
  const float*         ln_b  = (const float*)d_in[3];
  const float*         w_qkv = (const float*)d_in[4];
  const float*         b_qkv = (const float*)d_in[5];
  const float*         w_o   = (const float*)d_in[6];
  const float*         b_o   = (const float*)d_in[7];
  float*               out   = (float*)d_out;

  // Workspace carve-up (~131 MB total, 256B aligned slices).
  char* ws = (char*)d_ws;
  size_t off = 0;
  auto alloc = [&](size_t bytes) -> void* {
    void* p = ws + off;
    off = (off + bytes + 255) & ~(size_t)255;
    return p;
  };
  __bf16* h_bf    = (__bf16*)alloc((size_t)kNT * kD * 2);        // 16 MB
  __bf16* wqkvT   = (__bf16*)alloc((size_t)kQKV * kD * 2);       // 1.5 MB
  __bf16* woT     = (__bf16*)alloc((size_t)kD * kD * 2);         // 0.5 MB
  __bf16* qkv_bf  = (__bf16*)alloc((size_t)kNT * kQKV * 2);      // 48 MB
  __bf16* q_bf    = (__bf16*)alloc((size_t)kNT * kD * 2);        // 16 MB [B,H,T,64]
  __bf16* k_bf    = (__bf16*)alloc((size_t)kNT * kD * 2);        // 16 MB [B,H,T,64]
  __bf16* vT_bf   = (__bf16*)alloc((size_t)kNT * kD * 2);        // 16 MB [B,H,64,T]
  __bf16* ao_bf   = (__bf16*)alloc((size_t)kNT * kD * 2);        // 16 MB [B,T,512]

  prep_weights_kernel<<<(kQKV * kD + kD * kD) / 256, 256, 0, stream>>>(w_qkv, w_o, wqkvT, woT);
  ln_kernel<<<kNT, 256, 0, stream>>>(x, ln_w, ln_b, h_bf);
  qkv_gemm_kernel<<<dim3(kNT / 16, kQKV / 64), 32, 0, stream>>>(h_bf, wqkvT, b_qkv, qkv_bf);
  rope_kernel<<<(kB * kH * kT * 32) / 256, 256, 0, stream>>>(qkv_bf, q_bf, k_bf, vT_bf);
  attn_kernel<<<dim3(kT / 16, kB * kH), 32, 0, stream>>>(q_bf, k_bf, vT_bf, mask, ao_bf);
  out_gemm_kernel<<<dim3(kNT / 16, kD / 64), 32, 0, stream>>>(ao_bf, woT, b_o, out);
}
